// TransformerEncoder_85796266705287
// MI455X (gfx1250) — matrix operand order repaired and checked by reading, not compiled, not run on hardware
//
#include <hip/hip_runtime.h>
#include <hip/hip_bf16.h>
#include <stdint.h>

// ---------------------------------------------------------------------------
// CDNA5 WMMA (wave32): D(16x16 f32) = A(16x32 bf16) * B(32x16 bf16) + C
// ---------------------------------------------------------------------------
typedef __attribute__((ext_vector_type(16))) __bf16 v16bf;
typedef __attribute__((ext_vector_type(8)))  float  v8f;
typedef unsigned int u32x4 __attribute__((ext_vector_type(4)));
typedef int          i32x4 __attribute__((ext_vector_type(4)));
typedef int          i32x8 __attribute__((ext_vector_type(8)));

union FragU { v16bf v; uint4 q[2]; };

static __device__ __forceinline__ v8f wmma_bf16(v16bf a, v16bf b, v8f c) {
  return __builtin_amdgcn_wmma_f32_16x16x32_bf16(false, a, false, b, (short)0, c,
                                                 false, false);
}

// Problem constants
#define NROWS  16384      // b*n = 4*4096
#define DMODEL 512
#define NSEQ   4096
#define NBATCH 4
#define FFDIM  2048

// ---------------------------------------------------------------------------
// Tensor Data Mover: 2D tile (bf16) global -> LDS, one instruction per tile.
// D# built per CDNA5 ISA 8.3/8.4. Issued by one wave; tracked by TENSORcnt.
// ---------------------------------------------------------------------------
static __device__ __forceinline__ unsigned lds_off(const void* p) {
  // flat shared-aperture address: low 32 bits are the LDS byte offset
  return (unsigned)(uintptr_t)p;
}

static __device__ __forceinline__ void tdm_load_2d_bf16(
    unsigned ldsAddr, const void* gptr,
    unsigned tileW, unsigned tileH,      // elements / rows
    unsigned tensorW, unsigned tensorH,  // OOB extents (elements / rows)
    unsigned rowStride)                  // elements
{
  unsigned long long ga = (unsigned long long)(uintptr_t)gptr;
  u32x4 g0;
  g0[0] = 1u;                                    // count=1 (valid), user mode
  g0[1] = ldsAddr;                               // lds_addr [63:32]
  g0[2] = (unsigned)(ga & 0xFFFFFFFFu);          // global_addr[31:0]
  g0[3] = (unsigned)((ga >> 32) & 0x01FFFFFFu)   // global_addr[56:32]
        | (2u << 30);                            // type=2 ("image")
  i32x8 g1;
  g1[0] = (int)(1u << 16);                                  // data_size=1 (2B)
  g1[1] = (int)((tensorW & 0xFFFFu) << 16);                 // tensor_dim0[15:0] @63:48
  g1[2] = (int)(((tensorW >> 16) & 0xFFFFu)                 // tensor_dim0[31:16]
        |       ((tensorH & 0xFFFFu) << 16));               // tensor_dim1[15:0]
  g1[3] = (int)(((tensorH >> 16) & 0xFFFFu)                 // tensor_dim1[31:16]
        |       ((tileW & 0xFFFFu) << 16));                 // tile_dim0 @127:112
  g1[4] = (int)(tileH & 0xFFFFu);                           // tile_dim1; tile_dim2=0
  g1[5] = (int)rowStride;                                   // dim0_stride[31:0]
  g1[6] = 0;                                                // dim0_stride[47:32]=0
  g1[7] = 0;
  i32x4 z4 = {0, 0, 0, 0};
  i32x8 z8 = {0, 0, 0, 0, 0, 0, 0, 0};
  // amdgpu-toolchain (clang-23) 6-arg form: (g0, g1, g2, g3, g_extra, cpol)
  __builtin_amdgcn_tensor_load_to_lds(g0, g1, z4, z4, z8, 0);
}

// ---------------------------------------------------------------------------
// Cast fp32 -> bf16 (x4 vectorized)
// ---------------------------------------------------------------------------
struct alignas(8) B4 { __bf16 v[4]; };

__global__ void cast_f32_to_bf16(const float* __restrict__ in,
                                 __bf16* __restrict__ out, int n4) {
  int i = blockIdx.x * blockDim.x + threadIdx.x;
  if (i >= n4) return;
  float4 f = ((const float4*)in)[i];
  B4 r;
  r.v[0] = (__bf16)f.x; r.v[1] = (__bf16)f.y;
  r.v[2] = (__bf16)f.z; r.v[3] = (__bf16)f.w;
  ((B4*)out)[i] = r;
}

// ---------------------------------------------------------------------------
// Weight prep: W[K][N] f32 -> Wt[N][K] bf16 (transpose + convert), 64x64 tiles
// ---------------------------------------------------------------------------
__global__ __launch_bounds__(256) void prep_weight(const float* __restrict__ W,
                                                   __bf16* __restrict__ Wt,
                                                   int K, int N) {
  __shared__ __bf16 t[64][72];
  const int bn = blockIdx.x * 64, bk = blockIdx.y * 64;
  const int tid = threadIdx.x;
#pragma unroll
  for (int i = 0; i < 16; i++) {
    int idx = tid + 256 * i;
    int r = idx >> 6, c = idx & 63;               // r=k, c=n
    t[c][r] = (__bf16)W[(size_t)(bk + r) * N + bn + c];
  }
  __syncthreads();
#pragma unroll
  for (int i = 0; i < 16; i++) {
    int idx = tid + 256 * i;
    int r = idx >> 6, c = idx & 63;               // r=n, c=k
    Wt[(size_t)(bn + r) * K + bk + c] = t[r][c];
  }
}

// ---------------------------------------------------------------------------
// Batched bf16 transpose: in[z][R][C] -> out[z][C][R], 64x64 tiles (for V)
// ---------------------------------------------------------------------------
__global__ __launch_bounds__(256) void transpose_bf16(const __bf16* __restrict__ in,
                                                      __bf16* __restrict__ out,
                                                      int R, int C) {
  __shared__ __bf16 t[64][72];
  const size_t zoff = (size_t)blockIdx.z * R * C;
  const int bc = blockIdx.x * 64, br = blockIdx.y * 64;
  const int tid = threadIdx.x;
#pragma unroll
  for (int i = 0; i < 16; i++) {
    int idx = tid + 256 * i;
    int r = idx >> 6, c = idx & 63;               // r in R, c in C
    t[c][r] = in[zoff + (size_t)(br + r) * C + bc + c];
  }
  __syncthreads();
#pragma unroll
  for (int i = 0; i < 16; i++) {
    int idx = tid + 256 * i;
    int r = idx >> 6, c = idx & 63;               // r in C, c in R
    out[zoff + (size_t)(bc + r) * R + br + c] = t[r][c];
  }
}

// ---------------------------------------------------------------------------
// GEMM: Out[M,N] = A[M,K](bf16) @ Wt[N,K](bf16, pre-transposed) + bias
// 128x128 block tile, 8 waves x (32x64), K-step 32.
// Tiles staged by TDM (tensor_load_to_lds), double-buffered on TENSORcnt.
// ---------------------------------------------------------------------------
template <bool OBF, bool OF32, bool RELU>
__global__ __launch_bounds__(256) void gemm_tn(
    const __bf16* __restrict__ A, const __bf16* __restrict__ Bt,
    const float* __restrict__ bias, __bf16* __restrict__ Obf,
    float* __restrict__ Of32, int M, int N, int K) {
  __shared__ alignas(16) __bf16 sA[2][128][32];   // [m][k]
  __shared__ alignas(16) __bf16 sB[2][128][32];   // [n][k]

  const int tid  = threadIdx.x;
  const int wid  = tid >> 5;
  const int lane = tid & 31;
  const int l16  = lane & 15;
  const int half = lane >> 4;
  const int wy = wid >> 1;        // rows wy*32
  const int wx = wid & 1;         // cols wx*64
  const int bm = blockIdx.y * 128;
  const int bn = blockIdx.x * 128;

  const __bf16* Ablk = A  + (size_t)bm * K;
  const __bf16* Bblk = Bt + (size_t)bn * K;

  v8f acc[2][4];
#pragma unroll
  for (int a = 0; a < 2; a++)
#pragma unroll
    for (int b = 0; b < 4; b++) acc[a][b] = (v8f){};

  const int nk = K >> 5;
  if (wid == 0) {   // one TDM pair per tile, issued by wave 0
    tdm_load_2d_bf16(lds_off(&sA[0][0][0]), Ablk, 32, 128, K, 1u << 20, K);
    tdm_load_2d_bf16(lds_off(&sB[0][0][0]), Bblk, 32, 128, K, 1u << 20, K);
  }

  for (int t = 0; t < nk; t++) {
    const int cur = t & 1;
    if (wid == 0) {
      if (t + 1 < nk) {   // prefetch next tile into the other buffer
        tdm_load_2d_bf16(lds_off(&sA[cur ^ 1][0][0]), Ablk + (t + 1) * 32,
                         32, 128, K, 1u << 20, K);
        tdm_load_2d_bf16(lds_off(&sB[cur ^ 1][0][0]), Bblk + (t + 1) * 32,
                         32, 128, K, 1u << 20, K);
        __builtin_amdgcn_s_wait_tensorcnt(2);   // tile t complete
      } else {
        __builtin_amdgcn_s_wait_tensorcnt(0);
      }
    }
    __syncthreads();

    v16bf afrag[2], bfrag[4];
#pragma unroll
    for (int a = 0; a < 2; a++) {
      int r  = wy * 32 + a * 16 + l16;
      int ko = half * 8;            // A layout: K {0..7,16..23}/{8..15,24..31}
      FragU f;
      f.q[0] = *(const uint4*)(&sA[cur][r][ko]);
      f.q[1] = *(const uint4*)(&sA[cur][r][ko + 16]);
      afrag[a] = f.v;
    }
#pragma unroll
    for (int b = 0; b < 4; b++) {
      int c  = wx * 64 + b * 16 + l16;
      int ko = half * 8;            // B layout: K {0..7,16..23}/{8..15,24..31}
      FragU f;
      f.q[0] = *(const uint4*)(&sB[cur][c][ko]);
      f.q[1] = *(const uint4*)(&sB[cur][c][ko + 16]);
      bfrag[b] = f.v;
    }
#pragma unroll
    for (int a = 0; a < 2; a++)
#pragma unroll
      for (int b = 0; b < 4; b++)
        acc[a][b] = wmma_bf16(afrag[a], bfrag[b], acc[a][b]);
    __syncthreads();
  }

  // branch-free epilogue
#pragma unroll
  for (int a = 0; a < 2; a++)
#pragma unroll
    for (int b = 0; b < 4; b++) {
      int col = bn + wx * 64 + b * 16 + l16;
      float bv = bias[col];
#pragma unroll
      for (int v = 0; v < 8; v++) {
        int row = bm + wy * 32 + a * 16 + v + 8 * half;
        float val = acc[a][b][v] + bv;
        if (RELU) val = fmaxf(val, 0.f);
        if (OBF)  Obf [(size_t)row * N + col] = (__bf16)val;
        if (OF32) Of32[(size_t)row * N + col] = val;
      }
    }
}

// ---------------------------------------------------------------------------
// Flash attention over full HD=512 (no head split), scale 1/sqrt(64).
// Block: 64 query rows, 8 waves; streams 64-key blocks. Q/K/Vt tiles via TDM.
// Vt is the pre-transposed V: [D][NSEQ] per batch.
// ---------------------------------------------------------------------------
__global__ __launch_bounds__(256) void flash_attn(
    const __bf16* __restrict__ Q, const __bf16* __restrict__ Km,
    const __bf16* __restrict__ Vt, __bf16* __restrict__ O) {
  __shared__ alignas(16) __bf16 sQ[64][DMODEL];
  __shared__ alignas(16) __bf16 sK[64][DMODEL];
  __shared__ alignas(16) __bf16 sVt[DMODEL][64];
  __shared__ alignas(16) __bf16 sP[64][64];
  __shared__ float sM[64], sL[64], sScale[64];
  __shared__ float sPartM[64][2], sPartS[64][2];

  const int tid  = threadIdx.x;
  const int wid  = tid >> 5;
  const int lane = tid & 31;
  const int l16  = lane & 15;
  const int half = lane >> 4;
  const int wy   = wid >> 1;    // S phase rows wy*16
  const int wx   = wid & 1;     // S phase keys wx*32
  const int qbase = blockIdx.x * 64;
  const size_t boff = (size_t)blockIdx.y * NSEQ * DMODEL;
  const __bf16* Qb  = Q  + boff + (size_t)qbase * DMODEL;
  const __bf16* Kb0 = Km + boff;
  const __bf16* Vtb = Vt + boff;      // [DMODEL][NSEQ]

  if (wid == 0)
    tdm_load_2d_bf16(lds_off(&sQ[0][0]), Qb, DMODEL, 64, DMODEL, 64, DMODEL);
  if (tid < 64) { sM[tid] = -1e30f; sL[tid] = 0.f; }

  v8f acc[4][4];                      // O slice: 64 rows x 64 cols (wid*64)
#pragma unroll
  for (int a = 0; a < 4; a++)
#pragma unroll
    for (int b = 0; b < 4; b++) acc[a][b] = (v8f){};

  const float sscale = 0.125f;        // 1/sqrt(DK=64)

  for (int kb = 0; kb < NSEQ / 64; kb++) {
    __syncthreads();                  // prev PV done before overwriting sK/sVt
    if (wid == 0) {
      tdm_load_2d_bf16(lds_off(&sK[0][0]), Kb0 + (size_t)kb * 64 * DMODEL,
                       DMODEL, 64, DMODEL, 64, DMODEL);
      tdm_load_2d_bf16(lds_off(&sVt[0][0]), Vtb + (size_t)kb * 64,
                       64, DMODEL, NSEQ, DMODEL, NSEQ);
      __builtin_amdgcn_s_wait_tensorcnt(0);
    }
    __syncthreads();

    // ---- S = q @ k^T (16x32 per wave) ----
    v8f s[2];
    s[0] = (v8f){}; s[1] = (v8f){};
    for (int kk = 0; kk < DMODEL; kk += 32) {
      FragU fa;
      int r  = wy * 16 + l16;
      int ka = kk + half * 8;
      fa.q[0] = *(const uint4*)(&sQ[r][ka]);
      fa.q[1] = *(const uint4*)(&sQ[r][ka + 16]);
#pragma unroll
      for (int j = 0; j < 2; j++) {
        FragU fb;
        int c   = wx * 32 + j * 16 + l16;
        int kbo = kk + half * 8;
        fb.q[0] = *(const uint4*)(&sK[c][kbo]);
        fb.q[1] = *(const uint4*)(&sK[c][kbo + 16]);
        s[j] = wmma_bf16(fa.v, fb.v, s[j]);
      }
    }

    // ---- row max ----
    float mloc[8];
#pragma unroll
    for (int v = 0; v < 8; v++) {
      s[0][v] *= sscale; s[1][v] *= sscale;
      mloc[v] = fmaxf(s[0][v], s[1][v]);
    }
#pragma unroll
    for (int off = 8; off > 0; off >>= 1)
#pragma unroll
      for (int v = 0; v < 8; v++)
        mloc[v] = fmaxf(mloc[v], __shfl_xor(mloc[v], off, 32));
    if (l16 < 8) sPartM[wy * 16 + 8 * half + l16][wx] = mloc[l16];
    __syncthreads();
    if (tid < 64) {
      float mo = sM[tid];
      float mn = fmaxf(mo, fmaxf(sPartM[tid][0], sPartM[tid][1]));
      sScale[tid] = __expf(mo - mn);
      sM[tid] = mn;
    }
    __syncthreads();

    // ---- P = exp(S - m), row sums, write P (bf16) ----
    float mrow[8], ssum[8];
#pragma unroll
    for (int v = 0; v < 8; v++) mrow[v] = sM[wy * 16 + 8 * half + v];
#pragma unroll
    for (int v = 0; v < 8; v++) {
      float p0 = __expf(s[0][v] - mrow[v]);
      float p1 = __expf(s[1][v] - mrow[v]);
      int row = wy * 16 + 8 * half + v;
      sP[row][wx * 32 + l16]      = (__bf16)p0;
      sP[row][wx * 32 + 16 + l16] = (__bf16)p1;
      ssum[v] = p0 + p1;
    }
#pragma unroll
    for (int off = 8; off > 0; off >>= 1)
#pragma unroll
      for (int v = 0; v < 8; v++) ssum[v] += __shfl_xor(ssum[v], off, 32);
    if (l16 < 8) sPartS[wy * 16 + 8 * half + l16][wx] = ssum[l16];
    __syncthreads();
    if (tid < 64) sL[tid] = sL[tid] * sScale[tid] + sPartS[tid][0] + sPartS[tid][1];

    // ---- O = O*scale + P @ V (wave owns cols wid*64..+63) ----
    const int cw = wid * 64;
#pragma unroll
    for (int rt = 0; rt < 4; rt++) {
      float sc[8];
#pragma unroll
      for (int v = 0; v < 8; v++) sc[v] = sScale[rt * 16 + 8 * half + v];
#pragma unroll
      for (int ct = 0; ct < 4; ct++)
#pragma unroll
        for (int v = 0; v < 8; v++) acc[rt][ct][v] *= sc[v];
#pragma unroll
      for (int k2 = 0; k2 < 2; k2++) {
        FragU fa;
        int r  = rt * 16 + l16;
        int ka = k2 * 32 + half * 8;
        fa.q[0] = *(const uint4*)(&sP[r][ka]);
        fa.q[1] = *(const uint4*)(&sP[r][ka + 16]);
#pragma unroll
        for (int ct = 0; ct < 4; ct++) {
          FragU fb;
          int c   = cw + ct * 16 + l16;
          int kbo = k2 * 32 + half * 8;
          fb.q[0] = *(const uint4*)(&sVt[c][kbo]);
          fb.q[1] = *(const uint4*)(&sVt[c][kbo + 16]);
          acc[rt][ct] = wmma_bf16(fa.v, fb.v, acc[rt][ct]);
        }
      }
    }
  }
  __syncthreads();

  // ---- finalize: O /= l ----
  __bf16* Ob = O + boff + (size_t)qbase * DMODEL;
  const int cw = wid * 64;
#pragma unroll
  for (int rt = 0; rt < 4; rt++) {
    float li[8];
#pragma unroll
    for (int v = 0; v < 8; v++) li[v] = 1.f / sL[rt * 16 + 8 * half + v];
#pragma unroll
    for (int ct = 0; ct < 4; ct++) {
      int col = cw + ct * 16 + l16;
#pragma unroll
      for (int v = 0; v < 8; v++) {
        int row = rt * 16 + 8 * half + v;
        Ob[(size_t)row * DMODEL + col] = (__bf16)(acc[rt][ct][v] * li[v]);
      }
    }
  }
}

// ---------------------------------------------------------------------------
// y = LayerNorm(a + b) * g + beta ; writes f32 and/or bf16. One block per row.
// ---------------------------------------------------------------------------
__global__ __launch_bounds__(256) void add_ln(
    const float* __restrict__ A, const float* __restrict__ Bres,
    const float* __restrict__ g, const float* __restrict__ be,
    float* __restrict__ Yf, __bf16* __restrict__ Ybf) {
  __shared__ float wsum[8], wsum2[8];
  const int row = blockIdx.x;
  const int tid = threadIdx.x;
  const size_t base = (size_t)row * DMODEL;
  float x0 = A[base + tid]       + Bres[base + tid];
  float x1 = A[base + tid + 256] + Bres[base + tid + 256];
  float s  = x0 + x1;
  float s2 = x0 * x0 + x1 * x1;
#pragma unroll
  for (int off = 16; off > 0; off >>= 1) {
    s  += __shfl_xor(s,  off, 32);
    s2 += __shfl_xor(s2, off, 32);
  }
  if ((tid & 31) == 0) { wsum[tid >> 5] = s; wsum2[tid >> 5] = s2; }
  __syncthreads();
  if (tid == 0) {
    float t = 0.f, t2 = 0.f;
#pragma unroll
    for (int i = 0; i < 8; i++) { t += wsum[i]; t2 += wsum2[i]; }
    wsum[0] = t; wsum2[0] = t2;
  }
  __syncthreads();
  float mean = wsum[0] * (1.f / DMODEL);
  float var  = wsum2[0] * (1.f / DMODEL) - mean * mean;
  float inv  = rsqrtf(var + 1e-5f);
  float y0 = (x0 - mean) * inv * g[tid]       + be[tid];
  float y1 = (x1 - mean) * inv * g[tid + 256] + be[tid + 256];
  if (Yf)  { Yf[base + tid] = y0; Yf[base + tid + 256] = y1; }
  if (Ybf) { Ybf[base + tid] = (__bf16)y0; Ybf[base + tid + 256] = (__bf16)y1; }
}

// ---------------------------------------------------------------------------
// Launcher
// ---------------------------------------------------------------------------
extern "C" void kernel_launch(void* const* d_in, const int* in_sizes, int n_in,
                              void* d_out, int out_size, void* d_ws, size_t ws_size,
                              hipStream_t stream) {
  const float* x     = (const float*)d_in[0];
  const float* Wq_w  = (const float*)d_in[1];
  const float* Wq_b  = (const float*)d_in[2];
  const float* Wk_w  = (const float*)d_in[3];
  const float* Wk_b  = (const float*)d_in[4];
  const float* Wv_w  = (const float*)d_in[5];
  const float* Wv_b  = (const float*)d_in[6];
  const float* Wu_w  = (const float*)d_in[7];
  const float* Wu_b  = (const float*)d_in[8];
  const float* ln1_g = (const float*)d_in[9];
  const float* ln1_b = (const float*)d_in[10];
  const float* ff1_w = (const float*)d_in[11];
  const float* ff1_b = (const float*)d_in[12];
  const float* ff2_w = (const float*)d_in[13];
  const float* ff2_b = (const float*)d_in[14];
  const float* ln2_g = (const float*)d_in[15];
  const float* ln2_b = (const float*)d_in[16];

  const int    M  = NROWS;
  const size_t MD = (size_t)M * DMODEL;    // 8,388,608
  const size_t MF = (size_t)M * FFDIM;     // 33,554,432

  char* p = (char*)d_ws;
  __bf16* xb    = (__bf16*)p; p += MD * 2;
  __bf16* qb    = (__bf16*)p; p += MD * 2;
  __bf16* kb    = (__bf16*)p; p += MD * 2;
  __bf16* vb    = (__bf16*)p; p += MD * 2;
  __bf16* vtb   = (__bf16*)p; p += MD * 2;        // V transposed per batch
  __bf16* attnb = (__bf16*)p; p += MD * 2;
  __bf16* hb    = (__bf16*)p; p += MD * 2;
  __bf16* f1b   = (__bf16*)p; p += MF * 2;
  __bf16* wqt   = (__bf16*)p; p += (size_t)512 * 512 * 2;
  __bf16* wkt   = (__bf16*)p; p += (size_t)512 * 512 * 2;
  __bf16* wvt   = (__bf16*)p; p += (size_t)512 * 512 * 2;
  __bf16* wut   = (__bf16*)p; p += (size_t)512 * 512 * 2;
  __bf16* f1t   = (__bf16*)p; p += (size_t)2048 * 512 * 2;
  __bf16* f2t   = (__bf16*)p; p += (size_t)512 * 2048 * 2;
  float*  uni   = (float*)p;  p += MD * 4;
  float*  hf    = (float*)p;  p += MD * 4;
  float*  f2f   = (float*)p;  p += MD * 4;

  dim3 blk(256);

  cast_f32_to_bf16<<<dim3((unsigned)((MD / 4 + 255) / 256)), blk, 0, stream>>>(
      x, xb, (int)(MD / 4));

  // weights -> bf16, transposed to [N][K] for TDM-friendly GEMM B tiles
  prep_weight<<<dim3(8, 8),  blk, 0, stream>>>(Wq_w, wqt, 512, 512);
  prep_weight<<<dim3(8, 8),  blk, 0, stream>>>(Wk_w, wkt, 512, 512);
  prep_weight<<<dim3(8, 8),  blk, 0, stream>>>(Wv_w, wvt, 512, 512);
  prep_weight<<<dim3(8, 8),  blk, 0, stream>>>(Wu_w, wut, 512, 512);
  prep_weight<<<dim3(32, 8), blk, 0, stream>>>(ff1_w, f1t, 512, 2048);
  prep_weight<<<dim3(8, 32), blk, 0, stream>>>(ff2_w, f2t, 2048, 512);

  // Q, K, V projections (bf16 out)
  gemm_tn<true,  false, false><<<dim3(4, 128), blk, 0, stream>>>(xb, wqt, Wq_b, qb, nullptr, M, 512, 512);
  gemm_tn<true,  false, false><<<dim3(4, 128), blk, 0, stream>>>(xb, wkt, Wk_b, kb, nullptr, M, 512, 512);
  gemm_tn<true,  false, false><<<dim3(4, 128), blk, 0, stream>>>(xb, wvt, Wv_b, vb, nullptr, M, 512, 512);

  // V -> V^T per batch so the flash Vt tile is a plain TDM 2D tile
  transpose_bf16<<<dim3(8, 64, NBATCH), blk, 0, stream>>>(vb, vtb, NSEQ, DMODEL);

  // fused attention
  flash_attn<<<dim3(NSEQ / 64, NBATCH), blk, 0, stream>>>(qb, kb, vtb, attnb);

  // unify projection -> f32
  gemm_tn<false, true, false><<<dim3(4, 128), blk, 0, stream>>>(attnb, wut, Wu_b, nullptr, uni, M, 512, 512);

  // h = LN(x + unified): f32 + bf16
  add_ln<<<dim3(M), blk, 0, stream>>>(x, uni, ln1_g, ln1_b, hf, hb);

  // FF1(+ReLU) -> bf16, FF2 -> f32
  gemm_tn<true,  false, true ><<<dim3(16, 128), blk, 0, stream>>>(hb, f1t, ff1_b, f1b, nullptr, M, FFDIM, 512);
  gemm_tn<false, true, false><<<dim3(4, 128), blk, 0, stream>>>(f1b, f2t, ff2_b, nullptr, f2f, M, 512, FFDIM);

  // out = LN(h + ffwd)
  add_ln<<<dim3(M), blk, 0, stream>>>(hf, f2f, ln2_g, ln2_b, (float*)d_out, nullptr);
}
